// Head_34016140984659
// MI455X (gfx1250) — matrix operand order repaired and checked by reading, not compiled, not run on hardware
//
#include <hip/hip_runtime.h>
#include <hip/hip_bf16.h>

// ---------------------------------------------------------------------------
// Causal single-head attention for MI455X (gfx1250, wave32, WMMA).
//
// Roofline: ~21.5 GFLOP matrix math vs 72 MB HBM traffic (~3 us @ 23.3 TB/s)
// -> matrix-pipe bound.  Everything runs through v_wmma_f32_16x16x32_bf16
// (bf16 operands, f32 accumulate), flash-attention style so the 4M-entry
// score matrix never touches memory.  All bf16 intermediates (~45 MB) stay
// resident in the 192 MB L2.
//
// Software-pipelined GEMM loops; the pipeline is unrolled x2 so the
// current/next operand sets are statically renamed (no v_mov rotation
// copies, no WMMA->VALU hazard v_nops on the copies).
// ---------------------------------------------------------------------------

#define B_DIM 8
#define T_DIM 2048
#define E_DIM 1024
#define H_DIM 128

typedef __bf16 bf16;
typedef __attribute__((ext_vector_type(16))) __bf16 v16bf;
typedef __attribute__((ext_vector_type(8)))  __bf16 v8bf;
typedef __attribute__((ext_vector_type(8)))  float  v8f;

static __device__ __forceinline__ bf16 f2bf(float f) {
  unsigned u = __float_as_uint(f);
  u = (u + 0x7FFFu + ((u >> 16) & 1u)) >> 16;  // round-to-nearest-even
  unsigned short s = (unsigned short)u;
  bf16 r;
  __builtin_memcpy(&r, &s, sizeof(r));
  return r;
}

// A-operand 16x32 bf16 (interleaved K packing per ISA 7.12.2):
// caller passes &row[k0 + hi*8]; elems 0-7 <- [p..p+7], 8-15 <- [p+16..p+23].
static __device__ __forceinline__ v16bf load_opA(const bf16* p) {
  v8bf a0 = *(const v8bf*)p;
  v8bf a1 = *(const v8bf*)(p + 16);
  return __builtin_shufflevector(a0, a1, 0,1,2,3,4,5,6,7,8,9,10,11,12,13,14,15);
}

// B-operand 32x16 bf16 (interleaved K packing, same as A):
// caller passes &col[k0 + hi*8]; elems 0-7 <- [p..p+7], 8-15 <- [p+16..p+23].
static __device__ __forceinline__ v16bf load_opB(const bf16* p) {
  v8bf a0 = *(const v8bf*)p;
  v8bf a1 = *(const v8bf*)(p + 16);
  return __builtin_shufflevector(a0, a1, 0,1,2,3,4,5,6,7,8,9,10,11,12,13,14,15);
}

static __device__ __forceinline__ v8f wmma_bf16(v16bf a, v16bf b, v8f c) {
  return __builtin_amdgcn_wmma_f32_16x16x32_bf16(false, a, false, b, (short)0,
                                                 c, false, false);
}

// ---------------------------------------------------------------------------
// Kernel 1: x f32 -> bf16
// ---------------------------------------------------------------------------
__global__ void __launch_bounds__(256) cvt_x_kernel(const float* __restrict__ x,
                                                    bf16* __restrict__ xb, int n) {
  int i = (blockIdx.x * 256 + threadIdx.x) * 4;
  if (i >= n) return;
  float4 v = *(const float4*)(x + i);
  xb[i + 0] = f2bf(v.x);
  xb[i + 1] = f2bf(v.y);
  xb[i + 2] = f2bf(v.z);
  xb[i + 3] = f2bf(v.w);
}

// ---------------------------------------------------------------------------
// Kernel 2: W[E][H] f32 -> WT[3][H][E] bf16 (transposed so B-operand loads
// are contiguous in the reduction dim)
// ---------------------------------------------------------------------------
__global__ void __launch_bounds__(256) cvt_w_kernel(const float* __restrict__ Wk,
                                                    const float* __restrict__ Wq,
                                                    const float* __restrict__ Wv,
                                                    bf16* __restrict__ wt) {
  int tid = blockIdx.x * 256 + threadIdx.x;          // over 3*H*E
  int w = tid / (H_DIM * E_DIM);
  int r = tid % (H_DIM * E_DIM);
  int n = r / E_DIM;
  int e = r % E_DIM;
  const float* W = (w == 0) ? Wk : ((w == 1) ? Wq : Wv);
  wt[tid] = f2bf(W[e * H_DIM + n]);
}

// ---------------------------------------------------------------------------
// Kernel 3: fused QKV projection, software-pipelined (x2 unrolled pipeline).
// grid = M/32 row-tiles, 8 waves = 8 h-tiles over H=128.
// Each wave computes TWO 16x16 output tiles (rows r0..r0+31) for all three
// weights: one A pair + three B loads feed 6 WMMAs per K-step.
// K,Q stored [M][H]; V stored transposed [B][H][T] (contiguous C-layout store)
// ---------------------------------------------------------------------------
__global__ void __launch_bounds__(256) proj_kernel(const bf16* __restrict__ xb,
                                                   const bf16* __restrict__ wt,
                                                   bf16* __restrict__ kb,
                                                   bf16* __restrict__ qb,
                                                   bf16* __restrict__ vt) {
  const int lane = threadIdx.x & 31;
  const int wave = threadIdx.x >> 5;   // 0..7 -> n-tile over H
  const int l = lane & 15, hi = lane >> 4;
  const int r0 = blockIdx.x * 32;      // row tile over M = B*T
  const int n0 = wave * 16;

  const bf16* arow0 = xb + (size_t)(r0 + l) * E_DIM + hi * 8;
  const bf16* arow1 = arow0 + (size_t)16 * E_DIM;
  const bf16* bk = wt + (size_t)0 * H_DIM * E_DIM + (size_t)(n0 + l) * E_DIM + hi * 8;
  const bf16* bq = wt + (size_t)1 * H_DIM * E_DIM + (size_t)(n0 + l) * E_DIM + hi * 8;
  const bf16* bv = wt + (size_t)2 * H_DIM * E_DIM + (size_t)(n0 + l) * E_DIM + hi * 8;

  v8f ck0 = {}, cq0 = {}, cv0 = {}, ck1 = {}, cq1 = {}, cv1 = {};
  // pipeline registers: set X = even K-steps, set Y = odd K-steps
  v16bf A0x = load_opA(arow0),      A1x = load_opA(arow1);
  v16bf Bkx = load_opB(bk),         Bqx = load_opB(bq),  Bvx = load_opB(bv);
  v16bf A0y = load_opA(arow0 + 32), A1y = load_opA(arow1 + 32);
  v16bf Bky = load_opB(bk + 32),    Bqy = load_opB(bq + 32), Bvy = load_opB(bv + 32);

  // E_DIM/32 = 32 K-steps total; 15 double-steps in the loop + 1 drain pair.
#pragma unroll 1
  for (int e0 = 64; e0 < E_DIM; e0 += 64) {
    // consume set X, refill set X from e0
    ck0 = wmma_bf16(A0x, Bkx, ck0);
    ck1 = wmma_bf16(A1x, Bkx, ck1);
    cq0 = wmma_bf16(A0x, Bqx, cq0);
    cq1 = wmma_bf16(A1x, Bqx, cq1);
    cv0 = wmma_bf16(A0x, Bvx, cv0);
    cv1 = wmma_bf16(A1x, Bvx, cv1);
    A0x = load_opA(arow0 + e0);
    A1x = load_opA(arow1 + e0);
    Bkx = load_opB(bk + e0);
    Bqx = load_opB(bq + e0);
    Bvx = load_opB(bv + e0);
    // consume set Y, refill set Y from e0+32
    ck0 = wmma_bf16(A0y, Bky, ck0);
    ck1 = wmma_bf16(A1y, Bky, ck1);
    cq0 = wmma_bf16(A0y, Bqy, cq0);
    cq1 = wmma_bf16(A1y, Bqy, cq1);
    cv0 = wmma_bf16(A0y, Bvy, cv0);
    cv1 = wmma_bf16(A1y, Bvy, cv1);
    A0y = load_opA(arow0 + e0 + 32);
    A1y = load_opA(arow1 + e0 + 32);
    Bky = load_opB(bk + e0 + 32);
    Bqy = load_opB(bq + e0 + 32);
    Bvy = load_opB(bv + e0 + 32);
  }
  // drain both sets
  ck0 = wmma_bf16(A0x, Bkx, ck0);
  ck1 = wmma_bf16(A1x, Bkx, ck1);
  cq0 = wmma_bf16(A0x, Bqx, cq0);
  cq1 = wmma_bf16(A1x, Bqx, cq1);
  cv0 = wmma_bf16(A0x, Bvx, cv0);
  cv1 = wmma_bf16(A1x, Bvx, cv1);
  ck0 = wmma_bf16(A0y, Bky, ck0);
  ck1 = wmma_bf16(A1y, Bky, ck1);
  cq0 = wmma_bf16(A0y, Bqy, cq0);
  cq1 = wmma_bf16(A1y, Bqy, cq1);
  cv0 = wmma_bf16(A0y, Bvy, cv0);
  cv1 = wmma_bf16(A1y, Bvy, cv1);

  // K,Q: lane holds column h = n0+l, rows m = r0 + j + 8*hi (+16 for tile 1)
#pragma unroll
  for (int j = 0; j < 8; j++) {
    size_t m0 = (size_t)(r0 + j + 8 * hi);
    kb[m0 * H_DIM + n0 + l] = f2bf(ck0[j]);
    qb[m0 * H_DIM + n0 + l] = f2bf(cq0[j]);
    size_t m1 = m0 + 16;
    kb[m1 * H_DIM + n0 + l] = f2bf(ck1[j]);
    qb[m1 * H_DIM + n0 + l] = f2bf(cq1[j]);
  }
  // Vt[b][h][t]: lane's 8 values are consecutive t -> one 16B store per tile
  {
    int b = r0 / T_DIM;
    int tbase = (r0 % T_DIM) + 8 * hi;
    v8bf pv0, pv1;
#pragma unroll
    for (int j = 0; j < 8; j++) { pv0[j] = f2bf(cv0[j]); pv1[j] = f2bf(cv1[j]); }
    bf16* vrow = vt + ((size_t)(b * H_DIM + n0 + l)) * T_DIM;
    *(v8bf*)(vrow + tbase)      = pv0;
    *(v8bf*)(vrow + tbase + 16) = pv1;
  }
}

// ---------------------------------------------------------------------------
// Kernel 4: causal flash attention.  One wave per 16-query tile.
// Computes S^T = K*Q^T (lane = t => softmax reductions are lane-local and the
// exp'd P lands exactly in A-operand layout for P*V, no cross-lane shuffle).
// Operand loads are batched (one clause per phase) so WMMAs don't serialize
// behind loadcnt-0 waits; next chunk's K tiles are prefetched.
// ---------------------------------------------------------------------------
__global__ void __launch_bounds__(256) attn_kernel(const bf16* __restrict__ kb,
                                                   const bf16* __restrict__ qb,
                                                   const bf16* __restrict__ vt,
                                                   float* __restrict__ out) {
  const int lane = threadIdx.x & 31;
  const int wave = threadIdx.x >> 5;
  const int l = lane & 15, hi = lane >> 4;
  const int gq = blockIdx.x * 8 + wave;     // 0..1023
  const int b  = gq >> 7;
  const int t0 = (gq & 127) * 16;
  const float sc = 0.08838834764831845f;    // 1/sqrt(128)
  const int t_lane = t0 + l;

  // Preload Q as B-operand for all 4 h-chunks (K-dim = H = 128)
  const bf16* qrow = qb + ((size_t)(b * T_DIM + t0 + l)) * H_DIM + hi * 8;
  v16bf Bq[4];
#pragma unroll
  for (int c = 0; c < 4; c++) Bq[c] = load_opB(qrow + c * 32);

  const bf16* kbB = kb + (size_t)b * T_DIM * H_DIM;
  const bf16* vtB = vt + (size_t)b * H_DIM * T_DIM;

  v8f acc[8];
#pragma unroll
  for (int ht = 0; ht < 8; ht++) acc[ht] = (v8f){};
  float m_i = -3.0e38f, l_i = 0.0f;

  const int nchunk = (t0 + 47) / 32;        // cover s in [0, t0+16)
#pragma unroll 1
  for (int ch = 0; ch < nchunk; ch++) {
    const int s0 = ch * 32;
    const bf16* k0 = kbB + (size_t)(s0 + l) * H_DIM + hi * 8;
    const bf16* k1 = k0 + (size_t)16 * H_DIM;
    // prefetch next chunk's K rows (stays inside workspace; cheap cover)
    __builtin_prefetch(k0 + (size_t)32 * H_DIM, 0, 1);
    __builtin_prefetch(k1 + (size_t)32 * H_DIM, 0, 1);

    // ---- batched K operand loads, then 8 QK^T WMMAs ----
    v16bf Ka0[4], Ka1[4];
#pragma unroll
    for (int c = 0; c < 4; c++) Ka0[c] = load_opA(k0 + c * 32);
#pragma unroll
    for (int c = 0; c < 4; c++) Ka1[c] = load_opA(k1 + c * 32);
    v8f sa = {}, sb = {};
#pragma unroll
    for (int c = 0; c < 4; c++) {
      sa = wmma_bf16(Ka0[c], Bq[c], sa);
      sb = wmma_bf16(Ka1[c], Bq[c], sb);
    }

    // ---- scale + causal mask + chunk max (lane-local, then pair merge) ----
    float p0[8], p1[8], mloc = -3.0e38f;
#pragma unroll
    for (int j = 0; j < 8; j++) {
      int s_a = s0 + j + 8 * hi;
      float xa = sa[j] * sc;
      if (s_a > t_lane) xa = -3.0e38f;
      p0[j] = xa; mloc = fmaxf(mloc, xa);
      float xb2 = sb[j] * sc;
      if (s_a + 16 > t_lane) xb2 = -3.0e38f;
      p1[j] = xb2; mloc = fmaxf(mloc, xb2);
    }
    mloc = fmaxf(mloc, __shfl_xor(mloc, 16));
    float m_new = fmaxf(m_i, mloc);
    float alpha = __expf(m_i - m_new);

    // ---- exp, running sum, and pack P straight into A-operand layout ----
    float lsum = 0.0f;
    v16bf Pb;
#pragma unroll
    for (int j = 0; j < 8; j++) {
      float e0 = __expf(p0[j] - m_new); lsum += e0; Pb[j]     = f2bf(e0);
      float e1 = __expf(p1[j] - m_new); lsum += e1; Pb[j + 8] = f2bf(e1);
    }
    lsum += __shfl_xor(lsum, 16);
    l_i = l_i * alpha + lsum;
    m_i = m_new;
    // alpha per accumulator row t_local = j + 8*hi (broadcast via permlane)
    float aj[8];
#pragma unroll
    for (int j = 0; j < 8; j++) aj[j] = __shfl(alpha, j + 8 * hi);

    // ---- rescale O and accumulate P * V, V operands in batches of 4 ----
    v16bf Bv[4];
#pragma unroll
    for (int i = 0; i < 4; i++)
      Bv[i] = load_opB(vtB + (size_t)(i * 16 + l) * T_DIM + s0 + hi * 8);
#pragma unroll
    for (int i = 0; i < 4; i++) {
      v8f a = acc[i];
#pragma unroll
      for (int j = 0; j < 8; j++) a[j] *= aj[j];
      acc[i] = wmma_bf16(Pb, Bv[i], a);
    }
#pragma unroll
    for (int i = 0; i < 4; i++)
      Bv[i] = load_opB(vtB + (size_t)((i + 4) * 16 + l) * T_DIM + s0 + hi * 8);
#pragma unroll
    for (int i = 0; i < 4; i++) {
      v8f a = acc[i + 4];
#pragma unroll
      for (int j = 0; j < 8; j++) a[j] *= aj[j];
      acc[i + 4] = wmma_bf16(Pb, Bv[i], a);
    }
  }

  // ---- epilogue: out[b][t][h] = acc / l  (l broadcast per row) ----
  float linv = 1.0f / l_i;
  float lj[8];
#pragma unroll
  for (int j = 0; j < 8; j++) lj[j] = __shfl(linv, j + 8 * hi);
  float* ob = out + (size_t)b * T_DIM * H_DIM;
#pragma unroll
  for (int ht = 0; ht < 8; ht++) {
#pragma unroll
    for (int j = 0; j < 8; j++) {
      int t = t0 + j + 8 * hi;
      ob[(size_t)t * H_DIM + ht * 16 + l] = acc[ht][j] * lj[j];
    }
  }
}

// ---------------------------------------------------------------------------
extern "C" void kernel_launch(void* const* d_in, const int* in_sizes, int n_in,
                              void* d_out, int out_size, void* d_ws, size_t ws_size,
                              hipStream_t stream) {
  (void)in_sizes; (void)n_in; (void)out_size; (void)ws_size;
  const float* x  = (const float*)d_in[0];
  const float* Wk = (const float*)d_in[1];
  const float* Wq = (const float*)d_in[2];
  const float* Wv = (const float*)d_in[3];
  float* out = (float*)d_out;

  const size_t M = (size_t)B_DIM * T_DIM;  // 16384
  char* p = (char*)d_ws;                   // ~45 MB total workspace
  bf16* xb = (bf16*)p; p += M * E_DIM * sizeof(bf16);                 // 32 MB
  bf16* wt = (bf16*)p; p += (size_t)3 * H_DIM * E_DIM * sizeof(bf16); // 768 KB
  bf16* kb = (bf16*)p; p += M * H_DIM * sizeof(bf16);                 // 4 MB
  bf16* qb = (bf16*)p; p += M * H_DIM * sizeof(bf16);                 // 4 MB
  bf16* vt = (bf16*)p;                                                // 4 MB

  cvt_x_kernel<<<(int)(M * E_DIM / (256 * 4)), 256, 0, stream>>>(x, xb,
                                                                 (int)(M * E_DIM));
  cvt_w_kernel<<<3 * H_DIM * E_DIM / 256, 256, 0, stream>>>(Wk, Wq, Wv, wt);
  proj_kernel<<<(int)(M / 32), 256, 0, stream>>>(xb, wt, kb, qb, vt);
  attn_kernel<<<(int)(M / 16 / 8), 256, 0, stream>>>(kb, qb, vt, out);
}
